// SoftcbDiceLoss_78529182040561
// MI455X (gfx1250) — compile-verified, matrix operand order checked by reading, not run on hardware
//
#include <hip/hip_runtime.h>
#include <cstdint>

typedef __attribute__((ext_vector_type(16))) _Float16 v16h;
typedef __attribute__((ext_vector_type(8)))  float    v8f;

#define NB   4
#define H_   256
#define W_   256
#define HW   65536
#define NPIX (NB * HW)      // 262144
#define WPR  8              // u32 words per row
#define WPI  2048           // words per image
#define WPS  8192           // words per side (4 images)
#define FT   512            // threads in final reduction block (16 waves)

// ---------------- K1: pp = softmax([1-p, p])[1] = sigmoid(2*sigmoid(x)-1) ----------------
__global__ void k_pointwise(const float* __restrict__ y_pred, float* __restrict__ pp) {
  int i = blockIdx.x * blockDim.x + threadIdx.x;
  if (i < NPIX) {
    float p = 1.f / (1.f + expf(-y_pred[i]));
    pp[i] = 1.f / (1.f + expf(-(2.f * p - 1.f)));
  }
}

// ---------------- K2: bit-pack binary masks (side 0 = pred hard, side 1 = true) ----------
__global__ void k_pack(const float* __restrict__ pp, const int* __restrict__ yt,
                       unsigned* __restrict__ mbits) {
  int idx = blockIdx.x * blockDim.x + threadIdx.x;   // word id in [0, 2*WPS)
  if (idx >= 2 * WPS) return;
  int side = idx >> 13;
  int wi   = idx & (WPS - 1);
  int p0   = wi << 5;
  unsigned w = 0u;
  if (side == 0) {
    for (int i = 0; i < 32; ++i) w |= (pp[p0 + i] > 0.5f) ? (1u << i) : 0u;
  } else {
    for (int i = 0; i < 32; ++i) w |= (yt[p0 + i] > 0) ? (1u << i) : 0u;
  }
  mbits[idx] = w;
}

// ---------------- K3: bitwise soft_skel (11 openings on binary data) ---------------------
__device__ __forceinline__ unsigned erode_w(const unsigned* B, int r, int w) {
  unsigned c  = B[r * WPR + w];
  unsigned up = (r > 0)       ? B[(r - 1) * WPR + w] : 0xFFFFFFFFu;
  unsigned dn = (r < H_ - 1)  ? B[(r + 1) * WPR + w] : 0xFFFFFFFFu;
  unsigned li = (w > 0)       ? (B[r * WPR + w - 1] >> 31) : 1u;
  unsigned ri = (w < WPR - 1) ? (B[r * WPR + w + 1] & 1u)  : 1u;
  unsigned hl = (c << 1) | (li & 1u);
  unsigned hr = (c >> 1) | (ri << 31);
  return c & up & dn & hl & hr;   // min over 5-point cross (OOB treated as 1)
}
__device__ __forceinline__ unsigned dilrow_w(const unsigned* B, int r, int w) {
  if (r < 0 || r > H_ - 1) return 0u;
  unsigned c  = B[r * WPR + w];
  unsigned li = (w > 0)       ? (B[r * WPR + w - 1] >> 31) : 0u;
  unsigned ri = (w < WPR - 1) ? (B[r * WPR + w + 1] & 1u)  : 0u;
  return c | (c << 1) | (li & 1u) | (c >> 1) | (ri << 31);
}

__global__ void k_skel(const unsigned* __restrict__ mbits, unsigned* __restrict__ sbits) {
  __shared__ unsigned bufA[H_ * WPR];
  __shared__ unsigned bufB[H_ * WPR];
  int base = blockIdx.x * WPI;     // blockIdx = side*4 + b
  int r = threadIdx.x;             // one thread per row
  unsigned skel[WPR];
  for (int w = 0; w < WPR; ++w) { bufA[r * WPR + w] = mbits[base + r * WPR + w]; skel[w] = 0u; }
  unsigned* cur = bufA; unsigned* nxt = bufB;
  for (int t = 0; t <= 10; ++t) {          // skel |= cur & ~dilate(erode(cur)); cur = erode(cur)
    __syncthreads();
    unsigned e[WPR];
    for (int w = 0; w < WPR; ++w) e[w] = erode_w(cur, r, w);
    for (int w = 0; w < WPR; ++w) nxt[r * WPR + w] = e[w];
    __syncthreads();
    for (int w = 0; w < WPR; ++w) {
      unsigned o = dilrow_w(nxt, r - 1, w) | dilrow_w(nxt, r, w) | dilrow_w(nxt, r + 1, w);
      skel[w] |= cur[r * WPR + w] & ~o;
    }
    unsigned* tmp = cur; cur = nxt; nxt = tmp;
  }
  __syncthreads();
  for (int w = 0; w < WPR; ++w) sbits[base + r * WPR + w] = skel[w];
}

// ---------------- K4: horizontal distance scans -> g^2, stored transposed [sb][x][y] -----
__global__ void k_rowdist(const unsigned* __restrict__ mbits, float* __restrict__ g2T) {
  __shared__ float f[32][W_ + 1];
  int rg = blockIdx.x * 32 + threadIdx.x;  // [0, 2048): sb*256 + y
  int sb = rg >> 8;
  int y  = rg & 255;
  unsigned rw[WPR];
  for (int w = 0; w < WPR; ++w) rw[w] = mbits[sb * WPI + y * WPR + w];
  int t = threadIdx.x;
  float d = 512.f;                                   // BIG = H + W
  for (int x = 0; x < W_; ++x) {
    bool bg = !((rw[x >> 5] >> (x & 31)) & 1u);
    d = bg ? 0.f : fminf(d + 1.f, 512.f);
    f[t][x] = d;
  }
  d = 512.f;
  float* out = g2T + (size_t)sb * HW;
  for (int x = W_ - 1; x >= 0; --x) {
    bool bg = !((rw[x >> 5] >> (x & 31)) & 1u);
    d = bg ? 0.f : fminf(d + 1.f, 512.f);
    float g = fminf(d, f[t][x]);
    out[x * W_ + y] = g * g;
  }
}

// ---------------- K5: EDT lower envelope (async column stage into LDS) -------------------
__global__ void k_envelope(const float* __restrict__ g2T, const unsigned* __restrict__ mbits,
                           float* __restrict__ dist) {
  __shared__ float col[H_];
  int blk = blockIdx.x;            // [0, 2048): sb*256 + x
  int sb = blk >> 8;
  int x  = blk & 255;
  int y  = threadIdx.x;            // 256 threads, one output row each
  const float* src = g2T + (size_t)sb * HW + (size_t)x * W_;
  unsigned lds_addr = (unsigned)(uintptr_t)(&col[y]);
  unsigned voff = (unsigned)(y * 4);
  // CDNA5 async copy global -> LDS (ASYNCcnt path)
  asm volatile("global_load_async_to_lds_b32 %0, %1, %2"
               :: "v"(lds_addr), "v"(voff), "s"(src) : "memory");
  asm volatile("s_wait_asynccnt 0" ::: "memory");
  __syncthreads();
  float m = 3.4e38f;
  float yf = (float)y;
  #pragma unroll 8
  for (int yp = 0; yp < H_; ++yp) {
    float dy = yf - (float)yp;
    m = fminf(m, fmaf(dy, dy, col[yp]));
  }
  bool msk = (mbits[sb * WPI + y * WPR + (x >> 5)] >> (x & 31)) & 1u;
  dist[(size_t)sb * HW + (size_t)y * W_ + x] = msk ? sqrtf(m) : 0.f;  // masked distances
}

// ---------------- K6: per-(side,image) max/min of skel_radius ----------------------------
__global__ void k_stats(const float* __restrict__ pp, const float* __restrict__ dist,
                        const unsigned* __restrict__ sbits, float* __restrict__ stats) {
  __shared__ float smx[256], smn[256];
  int sb = blockIdx.x;             // side*4 + b
  int side = sb >> 2;
  int tid = threadIdx.x;
  float mx = 0.f, mn = 3.4e38f;
  for (int i = tid; i < HW; i += 256) {
    float dv = dist[(size_t)sb * HW + i];
    bool sk = (sbits[sb * WPI + (i >> 5)] >> (i & 31)) & 1u;
    if (side == 0) sk = sk && (pp[(sb & 3) * HW + i] > 0.5f);  // skel_pred_prob > 0.5
    float sr = sk ? dv : 0.f;
    mx = fmaxf(mx, sr);
    mn = fminf(mn, sr);
  }
  smx[tid] = mx; smn[tid] = mn;
  __syncthreads();
  for (int s = 128; s > 0; s >>= 1) {
    if (tid < s) { smx[tid] = fmaxf(smx[tid], smx[tid + s]); smn[tid] = fminf(smn[tid], smn[tid + s]); }
    __syncthreads();
  }
  if (tid == 0) { stats[sb * 2 + 0] = smx[0]; stats[sb * 2 + 1] = smn[0]; }
}

// ---------------- K7: fused weights + WMMA reduction + loss ------------------------------
// 512 threads (16 waves): larger per-wave VGPR budget keeps the 4x v16h staging and
// 4x v8f accumulators resident (no scratch spills). All divisions replaced by
// per-image reciprocals precomputed in LDS.
__global__ void __launch_bounds__(FT) k_final(
    const float* __restrict__ pp, const float* __restrict__ dist,
    const unsigned* __restrict__ mbits, const unsigned* __restrict__ sbits,
    const float* __restrict__ stats, float* __restrict__ out) {
  __shared__ float rmx[8], rmn[8], rin[8];   // max, min, 1/max per (side,b)
  __shared__ float red[4][FT];
  int tid = threadIdx.x;
  if (tid < 8) {
    float mx = fmaxf(stats[tid * 2 + 0], 1.f);
    rmx[tid] = mx;
    rmn[tid] = fmaxf(stats[tid * 2 + 1], 1.f);
    rin[tid] = 1.f / mx;                     // one divide per image, not per pixel
  }
  __syncthreads();
  v16h ones;
  #pragma unroll
  for (int i = 0; i < 16; ++i) ones[i] = (_Float16)1.0f;
  v8f c1 = {}, c2 = {}, c3 = {}, c4 = {};
  v16h b1, b2, b3, b4;
  for (int g = 0; g < NPIX / (16 * FT); ++g) {   // 32 groups * 16 slots * 512 threads = NPIX
    #pragma unroll
    for (int s = 0; s < 16; ++s) {
      int p = (g * 16 + s) * FT + tid;
      __builtin_prefetch(&dist[(p + FT) & (NPIX - 1)], 0, 1);
      int b  = p >> 16;
      int wi = p >> 5;
      unsigned bm = 1u << (p & 31);
      float ppv = pp[p];
      bool mt  = mbits[WPS + wi] & bm;     // true mask
      bool st  = sbits[WPS + wi] & bm;     // true skeleton
      bool sph = sbits[wi] & bm;           // pred skeleton (of hard)
      float dt = dist[NPIX + p];           // masked EDT, true side
      float dp = dist[p];                  // masked EDT, pred side
      float rmt = rmx[4 + b], rnt = rmn[4 + b], rit = rin[4 + b];
      float rmp = rmx[b],     rnp = rmn[b],     rip = rin[b];
      // true side (binary weights)
      float srt    = st ? dt : 0.f;
      float q_vl   = mt ? fminf(dt, rmt) * rit : 0.f;
      float q_slvl = mt ? srt * rit : 0.f;
      float q_sl   = st ? (rmt - srt + rnt) * rit : 0.f;
      // pred side (probabilistic weights)
      float spp = sph ? ppv : 0.f;         // skel_pred_prob
      bool  sp  = spp > 0.5f;
      float srp = sp ? dp : 0.f;
      float q_vp   = (fminf(dp, rmp) * rip) * ppv;
      float q_spvp = (srp * rip) * ppv;
      float q_sp   = (sp ? (rmp - srp + rnp) * rip : 0.f) * spp;
      float v1 = q_sp * q_vl;
      float v2 = ((q_spvp != 0.f) && (q_slvl == 0.f)) ? q_spvp * q_sp : q_slvl * q_sp;
      float v3 = q_sl * q_vp;
      float v4 = ((q_slvl != 0.f) && (q_spvp == 0.f)) ? q_slvl * q_sl : q_spvp * q_sl;
      b1[s] = (_Float16)v1; b2[s] = (_Float16)v2;
      b3[s] = (_Float16)v3; b4[s] = (_Float16)v4;
    }
    // ones-A WMMA: C[m,n] += sum_k B[k,n] -> layout-agnostic sum of all 512 B elements (x16 replication)
    c1 = __builtin_amdgcn_wmma_f32_16x16x32_f16(false, ones, false, b1, (short)0, c1, false, false);
    c2 = __builtin_amdgcn_wmma_f32_16x16x32_f16(false, ones, false, b2, (short)0, c2, false, false);
    c3 = __builtin_amdgcn_wmma_f32_16x16x32_f16(false, ones, false, b3, (short)0, c3, false, false);
    c4 = __builtin_amdgcn_wmma_f32_16x16x32_f16(false, ones, false, b4, (short)0, c4, false, false);
  }
  float l1 = 0.f, l2 = 0.f, l3 = 0.f, l4 = 0.f;
  #pragma unroll
  for (int i = 0; i < 8; ++i) { l1 += c1[i]; l2 += c2[i]; l3 += c3[i]; l4 += c4[i]; }
  red[0][tid] = l1; red[1][tid] = l2; red[2][tid] = l3; red[3][tid] = l4;
  __syncthreads();
  for (int s = FT / 2; s > 0; s >>= 1) {
    if (tid < s) for (int k = 0; k < 4; ++k) red[k][tid] += red[k][tid + s];
    __syncthreads();
  }
  if (tid == 0) {
    float S1 = red[0][0] / 16.f, S2 = red[1][0] / 16.f;   // /16: WMMA row replication
    float S3 = red[2][0] / 16.f, S4 = red[3][0] / 16.f;
    float wtp = (S1 + 1.f) / (S2 + 1.f);
    float wts = (S3 + 1.f) / (S4 + 1.f);
    out[0] = 1.f - 2.f * (wtp * wts) / (wtp + wts);
  }
}

// ---------------- launch ----------------------------------------------------------------
extern "C" void kernel_launch(void* const* d_in, const int* in_sizes, int n_in,
                              void* d_out, int out_size, void* d_ws, size_t ws_size,
                              hipStream_t stream) {
  const float* y_pred = (const float*)d_in[0];
  const int*   y_true = (const int*)d_in[1];
  float* out = (float*)d_out;

  // workspace layout (~5.4 MB)
  float*    pp    = (float*)d_ws;            // NPIX
  float*    dist  = pp + NPIX;               // 2*NPIX (side 0 = pred, side 1 = true)
  float*    g2T   = dist + 2 * NPIX;         // 2*NPIX, transposed [sb][x][y]
  unsigned* mbits = (unsigned*)(g2T + 2 * NPIX);  // 2*WPS
  unsigned* sbits = mbits + 2 * WPS;              // 2*WPS
  float*    stats = (float*)(sbits + 2 * WPS);    // 16

  k_pointwise<<<NPIX / 256, 256, 0, stream>>>(y_pred, pp);
  k_pack     <<<(2 * WPS) / 256, 256, 0, stream>>>(pp, y_true, mbits);
  k_skel     <<<8, 256, 0, stream>>>(mbits, sbits);
  k_rowdist  <<<64, 32, 0, stream>>>(mbits, g2T);
  k_envelope <<<2048, 256, 0, stream>>>(g2T, mbits, dist);
  k_stats    <<<8, 256, 0, stream>>>(pp, dist, sbits, stats);
  k_final    <<<1, FT, 0, stream>>>(pp, dist, mbits, sbits, stats, out);
}
